// SelfAttentionHead_40492951666987
// MI455X (gfx1250) — matrix operand order repaired and checked by reading, not compiled, not run on hardware
//
#include <hip/hip_runtime.h>

// Self-attention head, fp32 reference, computed via bf16 WMMA with f32 accum.
// B=4, T=4096, N_EMBD=2048, HEAD=128.

#define B_  4
#define T_  4096
#define NE_ 2048
#define H_  128

typedef __attribute__((ext_vector_type(16))) __bf16 v16bf;
typedef __attribute__((ext_vector_type(2)))  __bf16 v2bf;
typedef __attribute__((ext_vector_type(8)))  float  v8f;

union ABf {
  v16bf v;
  unsigned int u[8];
  unsigned short h[16];
};

// Native conversions -> hardware v_cvt_pk_bf16_f32 / v_cvt_bf16_f32.
__device__ __forceinline__ unsigned int packbf(float lo, float hi) {
  v2bf r;
  r[0] = (__bf16)lo;
  r[1] = (__bf16)hi;
  return __builtin_bit_cast(unsigned int, r);
}
__device__ __forceinline__ unsigned short bf16bits(float f) {
  __bf16 b = (__bf16)f;
  return __builtin_bit_cast(unsigned short, b);
}

__device__ __forceinline__ v8f wmma_bf16(v16bf a, v16bf b, v8f c) {
  // (neg_a, A, neg_b, B, c_mod, C, reuse_a, reuse_b)
  return __builtin_amdgcn_wmma_f32_16x16x32_bf16(false, a, false, b,
                                                 (short)0, c, false, false);
}

// A-operand (16x32 bf16, MxK): lane m=lane&15, half=lane>>4;
// VGPR v holds K pair k0,k0+1 with k0 = 2v + 16*half.
__device__ __forceinline__ int a_k0(int v, int half) {
  return 2 * v + 16 * half;
}

// ---------------------------------------------------------------------------
// Kernel 1: QKV projection. One wave per 16(M)x16(N) tile; A (x tile) shared
// by three WMMAs per K-step. Outputs bf16: q,k -> [b,t,h]; v -> [b,h,t] (T).
// ---------------------------------------------------------------------------
__global__ void __launch_bounds__(32) qkv_proj_kernel(
    const float* __restrict__ x,  const float* __restrict__ Wq,
    const float* __restrict__ Wk, const float* __restrict__ Wv,
    unsigned short* __restrict__ qb, unsigned short* __restrict__ kbuf,
    unsigned short* __restrict__ vtb)
{
  const int lane = threadIdx.x;
  const int m    = lane & 15;
  const int half = lane >> 4;
  const int mt   = blockIdx.x;            // 0 .. B*T/16-1
  const int b    = mt >> 8;               // 256 M-tiles per batch
  const int t0   = (mt & 255) << 4;
  const int h0   = blockIdx.y << 4;       // N tile

  const float* xrow = x + ((size_t)b * T_ + t0 + m) * NE_;
  v8f cq = {}, ck = {}, cv = {};

  for (int kb = 0; kb < NE_; kb += 32) {
    // prefetch next x chunk for this row (streams from HBM once, then L2)
    if (kb + 32 < NE_) __builtin_prefetch(xrow + kb + 32 + 8 * half, 0, 3);

    ABf a;
#pragma unroll
    for (int v = 0; v < 8; ++v) {
      const float2 xx = *(const float2*)(xrow + kb + a_k0(v, half));
      a.u[v] = packbf(xx.x, xx.y);
    }
    ABf bq, bk, bv;
#pragma unroll
    for (int v = 0; v < 8; ++v) {
      // B-operand (32x16 KxN): VGPR v holds K pair (2v,2v+1)+16*half, col n=m
      const int kr = kb + 2 * v + 16 * half;
      const size_t o0 = (size_t)kr * H_ + h0 + m;
      bq.u[v] = packbf(Wq[o0], Wq[o0 + H_]);
      bk.u[v] = packbf(Wk[o0], Wk[o0 + H_]);
      bv.u[v] = packbf(Wv[o0], Wv[o0 + H_]);
    }
    cq = wmma_bf16(a.v, bq.v, cq);
    ck = wmma_bf16(a.v, bk.v, ck);
    cv = wmma_bf16(a.v, bv.v, cv);
  }

  // C/D layout: VGPR v holds row (v + 8*half), column = m
#pragma unroll
  for (int v = 0; v < 8; ++v) {
    const int row = v + 8 * half;
    const size_t qi = ((size_t)b * T_ + t0 + row) * H_ + h0 + m;
    qb[qi]   = bf16bits(cq[v]);
    kbuf[qi] = bf16bits(ck[v]);
    const size_t vi = ((size_t)b * H_ + h0 + m) * T_ + t0 + row;  // transposed
    vtb[vi] = bf16bits(cv[v]);
  }
}

// ---------------------------------------------------------------------------
// Kernel 2: causal flash attention. One wave per (batch, 16-query tile).
// ---------------------------------------------------------------------------
__global__ void __launch_bounds__(32) attn_kernel(
    const unsigned short* __restrict__ qb,
    const unsigned short* __restrict__ kbuf,
    const unsigned short* __restrict__ vtb,
    float* __restrict__ out)
{
  __shared__ unsigned short Plds[16 * 32];  // P tile staging (C-layout -> A-layout)
  const int lane = threadIdx.x;
  const int m    = lane & 15;
  const int half = lane >> 4;
  const int b    = blockIdx.y;
  const int t0   = blockIdx.x << 4;
  const float scale = 0.08838834764831845f;  // 1/sqrt(128)

  // Q tile as 4 pre-packed A-operands (K = 128 in 4 chunks of 32)
  ABf qa[4];
  const unsigned short* qrow = qb + ((size_t)b * T_ + t0 + m) * H_;
#pragma unroll
  for (int j = 0; j < 4; ++j)
#pragma unroll
    for (int v = 0; v < 8; ++v)
      qa[j].u[v] = *(const unsigned int*)(qrow + 32 * j + a_k0(v, half));

  v8f o[8] = {};                     // 16 rows x 128 cols of O, f32
  float mrow[8], lrow[8];
#pragma unroll
  for (int v = 0; v < 8; ++v) { mrow[v] = -__builtin_inff(); lrow[v] = 0.0f; }

  const unsigned short* vtall = vtb + (size_t)b * H_ * T_;

  for (int s0 = 0; s0 < t0 + 16; s0 += 32) {   // 32-key blocks, causal range
    const unsigned short* kbase = kbuf + ((size_t)b * T_ + s0) * H_;

    // prefetch next key block (K rows are contiguous; V^T rows are strided)
    if (s0 + 32 < t0 + 16) {
      __builtin_prefetch(kbase + 32 * H_ + lane * 128, 0, 3);       // 8KB K blk
      __builtin_prefetch(vtall + (size_t)lane       * T_ + s0 + 32, 0, 3);
      __builtin_prefetch(vtall + (size_t)(lane + 32) * T_ + s0 + 32, 0, 3);
      __builtin_prefetch(vtall + (size_t)(lane + 64) * T_ + s0 + 32, 0, 3);
      __builtin_prefetch(vtall + (size_t)(lane + 96) * T_ + s0 + 32, 0, 3);
    }

    // ---- scores = Q . K^T : two 16x16 tiles (keys s0..s0+15, s0+16..s0+31)
    v8f acc0 = {}, acc1 = {};
#pragma unroll
    for (int j = 0; j < 4; ++j) {
      ABf b0, b1;
#pragma unroll
      for (int v = 0; v < 8; ++v) {
        const int hh = 32 * j + 2 * v + 16 * half;   // B K-dim = head dim
        b0.u[v] = *(const unsigned int*)(kbase + (size_t)m        * H_ + hh);
        b1.u[v] = *(const unsigned int*)(kbase + (size_t)(m + 16) * H_ + hh);
      }
      acc0 = wmma_bf16(qa[j].v, b0.v, acc0);
      acc1 = wmma_bf16(qa[j].v, b1.v, acc1);
    }

    // ---- online softmax (rows live in (v,half); cols across 16 lanes)
    const bool diag = (s0 + 31 > t0);
    float alpha[8];
#pragma unroll
    for (int v = 0; v < 8; ++v) {
      const int trow = t0 + v + 8 * half;
      float s0v = acc0[v] * scale;
      float s1v = acc1[v] * scale;
      if (diag) {
        if (s0 + m      > trow) s0v = -__builtin_inff();
        if (s0 + 16 + m > trow) s1v = -__builtin_inff();
      }
      float mb = fmaxf(s0v, s1v);
      mb = fmaxf(mb, __shfl_xor(mb, 1));
      mb = fmaxf(mb, __shfl_xor(mb, 2));
      mb = fmaxf(mb, __shfl_xor(mb, 4));
      mb = fmaxf(mb, __shfl_xor(mb, 8));
      const float mnew = fmaxf(mrow[v], mb);
      const float al   = __expf(mrow[v] - mnew);
      const float p0   = __expf(s0v - mnew);
      const float p1   = __expf(s1v - mnew);
      Plds[(v + 8 * half) * 32 + m]      = bf16bits(p0);
      Plds[(v + 8 * half) * 32 + 16 + m] = bf16bits(p1);
      float rs = p0 + p1;
      rs += __shfl_xor(rs, 1);
      rs += __shfl_xor(rs, 2);
      rs += __shfl_xor(rs, 4);
      rs += __shfl_xor(rs, 8);
      lrow[v]  = lrow[v] * al + rs;
      mrow[v]  = mnew;
      alpha[v] = al;
    }
#pragma unroll
    for (int nt = 0; nt < 8; ++nt)
#pragma unroll
      for (int v = 0; v < 8; ++v) o[nt][v] *= alpha[v];

    asm volatile("s_wait_dscnt 0" ::: "memory");  // P stores visible to reads

    // ---- read P back in A-layout (K = 32 keys)
    ABf pa;
#pragma unroll
    for (int v = 0; v < 8; ++v)
      pa.u[v] = *(const unsigned int*)&Plds[m * 32 + a_k0(v, half)];

    // ---- O += P . V ; V^T layout makes K-dim pairs contiguous dwords
    const unsigned short* vtbase = vtall + s0;
#pragma unroll
    for (int nt = 0; nt < 8; ++nt) {
      ABf bv;
#pragma unroll
      for (int v = 0; v < 8; ++v) {
        const int key = 2 * v + 16 * half;
        bv.u[v] = *(const unsigned int*)(vtbase + (size_t)(16 * nt + m) * T_ + key);
      }
      o[nt] = wmma_bf16(pa.v, bv.v, o[nt]);
    }
  }

  // ---- normalize and store fp32
  float rl[8];
#pragma unroll
  for (int v = 0; v < 8; ++v) rl[v] = 1.0f / lrow[v];
#pragma unroll
  for (int nt = 0; nt < 8; ++nt)
#pragma unroll
    for (int v = 0; v < 8; ++v) {
      const size_t oi = ((size_t)b * T_ + t0 + v + 8 * half) * H_ + 16 * nt + m;
      out[oi] = o[nt][v] * rl[v];
    }
}

extern "C" void kernel_launch(void* const* d_in, const int* in_sizes, int n_in,
                              void* d_out, int out_size, void* d_ws, size_t ws_size,
                              hipStream_t stream) {
  const float* x  = (const float*)d_in[0];
  const float* Wq = (const float*)d_in[1];
  const float* Wk = (const float*)d_in[2];
  const float* Wv = (const float*)d_in[3];
  float* out = (float*)d_out;

  // workspace: 3 bf16 buffers of B*T*H (12 MB total)
  unsigned short* qb   = (unsigned short*)d_ws;
  unsigned short* kbuf = qb   + (size_t)B_ * T_ * H_;
  unsigned short* vtb  = kbuf + (size_t)B_ * T_ * H_;

  dim3 g1(B_ * T_ / 16, H_ / 16);
  qkv_proj_kernel<<<g1, 32, 0, stream>>>(x, Wq, Wk, Wv, qb, kbuf, vtb);

  dim3 g2(T_ / 16, B_);
  attn_kernel<<<g2, 32, 0, stream>>>(qb, kbuf, vtb, out);
}
